// LSHSelfAttention_24361054503215
// MI455X (gfx1250) — matrix operand order repaired and checked by reading, not compiled, not run on hardware
//
#include <hip/hip_runtime.h>

// ---------------- problem constants (match reference config) ----------------
#define B_      2
#define S_      4096
#define HID_    1024
#define H_      16
#define D_      64
#define NH_     4
#define NBUCK_  128
#define CHUNK_  64
#define NCHUNK_ (NH_ * S_ / CHUNK_)   // 256
#define TOT_    (NH_ * S_)            // 16384 per (b,h)

typedef float  f32x4 __attribute__((ext_vector_type(4)));
typedef __bf16 v16bf __attribute__((ext_vector_type(16)));
typedef float  v8f   __attribute__((ext_vector_type(8)));

union Frag16 {
    f32x4  q[2];
    v16bf  v;
    __bf16 h[16];
};

// D = A(16x32) * B(32x16) + C, bf16 inputs, f32 accum
__device__ __forceinline__ v8f wmma_bf16(v16bf a, v16bf b, v8f c) {
    return __builtin_amdgcn_wmma_f32_16x16x32_bf16(false, a, false, b,
                                                   (short)0, c, false, false);
}

// A fragment (16x32 bf16): lane half `hi` holds K = {hi*8..hi*8+7} U {16+hi*8..+7}
// from a contiguous row-major bf16 row.
__device__ __forceinline__ v16bf loadA_bf16(const __bf16* rowPtr, int kbase, int hi) {
    Frag16 f;
    f.q[0] = *(const f32x4*)(rowPtr + kbase + hi * 8);
    f.q[1] = *(const f32x4*)(rowPtr + kbase + 16 + hi * 8);
    return f.v;
}

// B fragment (32x16 bf16): lane half `hi` holds K = {hi*8..hi*8+7} U {16+hi*8..+7} of column n,
// sourced from a row-major [n][k] tile (so contiguous in k).
__device__ __forceinline__ v16bf loadB_rowk(const __bf16* rowPtr, int kbase, int hi) {
    Frag16 f;
    f.q[0] = *(const f32x4*)(rowPtr + kbase + hi * 8);
    f.q[1] = *(const f32x4*)(rowPtr + kbase + 16 + hi * 8);
    return f.v;
}

// A fragment from f32 memory with on-the-fly bf16 conversion
__device__ __forceinline__ v16bf loadA_f32(const float* rowPtr, int kbase, int hi) {
    f32x4 a0 = *(const f32x4*)(rowPtr + kbase + hi * 8);
    f32x4 a1 = *(const f32x4*)(rowPtr + kbase + hi * 8 + 4);
    f32x4 a2 = *(const f32x4*)(rowPtr + kbase + 16 + hi * 8);
    f32x4 a3 = *(const f32x4*)(rowPtr + kbase + 16 + hi * 8 + 4);
    Frag16 f;
#pragma unroll
    for (int i = 0; i < 4; ++i) {
        f.h[i]      = (__bf16)a0[i];
        f.h[4 + i]  = (__bf16)a1[i];
        f.h[8 + i]  = (__bf16)a2[i];
        f.h[12 + i] = (__bf16)a3[i];
    }
    return f.v;
}

// ---------------------------------------------------------------------------
// K1: projections  qk = hidden @ Wqk, v = hidden @ Wv  -> bf16 [b,h,s,d]
// one wave computes a 64x16 strip (4 M-tiles share one B fragment)
// ---------------------------------------------------------------------------
__global__ void proj_gemm(const float* __restrict__ hidden,
                          const float* __restrict__ wqk,
                          const float* __restrict__ wv,
                          __bf16* __restrict__ qkb,
                          __bf16* __restrict__ vb) {
    const int lane = threadIdx.x & 31;
    const int hi   = lane >> 4;
    const int ln   = lane & 15;
    const int nt   = blockIdx.x / 128;   // 0..63  (16-wide N tile)
    const int mt4  = blockIdx.x % 128;   // 0..127 (64-row M strip)
    const float* w  = (blockIdx.y == 0) ? wqk : wv;
    __bf16* outp    = (blockIdx.y == 0) ? qkb : vb;
    const int col   = nt * 16 + ln;

    v8f acc[4];
#pragma unroll
    for (int s = 0; s < 4; ++s) acc[s] = {};

    for (int ks = 0; ks < HID_; ks += 32) {
        Frag16 fb;
#pragma unroll
        for (int e = 0; e < 16; ++e) {
            int k = ks + hi * 8 + (e & 7) + ((e >> 3) << 4);
            fb.h[e] = (__bf16)w[k * HID_ + col];
        }
        v16bf bfrag = fb.v;
#pragma unroll
        for (int sub = 0; sub < 4; ++sub) {
            int m = mt4 * 64 + sub * 16 + ln;  // row in [0, B*S)
            v16bf afrag = loadA_f32(hidden + (size_t)m * HID_, ks, hi);
            acc[sub] = wmma_bf16(afrag, bfrag, acc[sub]);
        }
    }
    const int h = col >> 6, d = col & 63;
#pragma unroll
    for (int sub = 0; sub < 4; ++sub) {
#pragma unroll
        for (int r = 0; r < 8; ++r) {
            int m = mt4 * 64 + sub * 16 + r + 8 * hi;
            int b = m >> 12, s = m & (S_ - 1);
            outp[(((size_t)(b * H_ + h)) * S_ + s) * D_ + d] = (__bf16)acc[sub][r];
        }
    }
}

// ---------------------------------------------------------------------------
// K2: LSH hashing: rotated = qk . rotations, bucket = argmax(cat(r,-r)) + n*128
// ---------------------------------------------------------------------------
__global__ void lsh_hash(const __bf16* __restrict__ qkb,
                         const float* __restrict__ rot,
                         int* __restrict__ buckets) {
    const int s = blockIdx.x * 64 + threadIdx.x;
    const int h = blockIdx.y, b = blockIdx.z;
    const __bf16* row = qkb + (((size_t)(b * H_ + h)) * S_ + s) * D_;
    const int outBase = (b * H_ + h) * TOT_;

    for (int n = 0; n < NH_; ++n) {
        float acc[64];
#pragma unroll
        for (int r = 0; r < 64; ++r) acc[r] = 0.f;
        for (int d = 0; d < D_; ++d) {
            float qv = (float)row[d];
            const f32x4* rp = (const f32x4*)(rot + (size_t)(((h * D_ + d) * NH_ + n) << 6));
#pragma unroll
            for (int r4 = 0; r4 < 16; ++r4) {
                f32x4 rv = rp[r4];
                acc[r4 * 4 + 0] += qv * rv[0];
                acc[r4 * 4 + 1] += qv * rv[1];
                acc[r4 * 4 + 2] += qv * rv[2];
                acc[r4 * 4 + 3] += qv * rv[3];
            }
        }
        float best = acc[0];
        int   bi   = 0;
#pragma unroll
        for (int j = 1; j < 128; ++j) {
            float v = (j < 64) ? acc[j] : -acc[j - 64];
            if (v > best) { best = v; bi = j; }
        }
        buckets[outBase + n * S_ + s] = bi + n * NBUCK_;
    }
}

// ---------------------------------------------------------------------------
// K3: stable counting sort per (b,h). Key = (bucket, s), ties in n ascending:
// identical to argsort(S*bucket + s) with stable j order.
// ---------------------------------------------------------------------------
__global__ void bucket_sort(const int* __restrict__ buckets,
                            int* __restrict__ sidx) {
    const int bh   = blockIdx.x;
    const int tid  = threadIdx.x;       // 64 threads
    const int base = bh * TOT_;
    __shared__ int cnt[512];
    __shared__ int pref[512];
    __shared__ int runl[64][8];

#pragma unroll
    for (int i = 0; i < 8; ++i) { cnt[tid + i * 64] = 0; runl[tid][i] = 0; }
    __syncthreads();

    for (int t = tid; t < TOT_; t += 64)
        atomicAdd(&cnt[buckets[base + t]], 1);
    __syncthreads();

    if (tid == 0) {
        int run = 0;
        for (int i = 0; i < 512; ++i) { pref[i] = run; run += cnt[i]; }
    }
    __syncthreads();

    // traversal in (s asc, n asc) order; lane owns buckets with (bkt & 63)==tid
    for (int t = 0; t < TOT_; ++t) {
        int s  = t >> 2;
        int n  = t & 3;
        int j  = n * S_ + s;
        int bk = buckets[base + j];
        if ((bk & 63) == tid) {
            int pos = pref[bk] + runl[tid][bk >> 6]++;
            sidx[base + pos] = j;
        }
    }
}

// ---------------------------------------------------------------------------
// K4: chunked attention. One 128-thread block per (chunk, h, b).
// ---------------------------------------------------------------------------
__global__ void lsh_attn(const __bf16* __restrict__ qkb,
                         const __bf16* __restrict__ vb,
                         const int* __restrict__ sidx,
                         __bf16* __restrict__ outr,
                         float* __restrict__ logitsg) {
    __shared__ __align__(16) __bf16 Qs[64][64];     //  8 KB (raw queries)
    __shared__ __align__(16) __bf16 Ks[128][64];    // 16 KB (normalized keys)
    __shared__ __align__(16) __bf16 Vt[64][128];    // 16 KB (V transposed, d-major)
    __shared__ __align__(16) __bf16 Ps[64][128];    // 16 KB (probs)
    __shared__ int KIdx[128];
    __shared__ int QIdx[64];
    __shared__ int QJ[64];

    const int tid = threadIdx.x;
    const int c = blockIdx.x, h = blockIdx.y, b = blockIdx.z;
    const int bh = b * H_ + h;
    const size_t headOff = (size_t)bh * S_ * D_;
    const int sbase = bh * TOT_;

    // ---- gather Q/K/V rows (kv window = [chunk c-1, chunk c]) ----
    {
        const int rkv     = tid;  // 0..127
        const int kvchunk = (rkv < 64) ? ((c + NCHUNK_ - 1) % NCHUNK_) : c;
        const int pos     = kvchunk * CHUNK_ + (rkv & 63);
        const int j       = sidx[sbase + pos];
        const int s       = j & (S_ - 1);

        const __bf16* qrow = qkb + headOff + (size_t)s * D_;
        f32x4 buf4[8];
        const f32x4* qp = (const f32x4*)qrow;
#pragma unroll
        for (int i = 0; i < 8; ++i) buf4[i] = qp[i];
        const __bf16* buf = (const __bf16*)buf4;

        float ss = 0.f;
#pragma unroll
        for (int d = 0; d < 64; ++d) { float f = (float)buf[d]; ss += f * f; }
        const float scale = rsqrtf(ss * (1.f / 64.f) + 1e-6f) * 0.125f; // rms norm * d^-0.5
#pragma unroll
        for (int d = 0; d < 64; ++d) Ks[rkv][d] = (__bf16)((float)buf[d] * scale);
        KIdx[rkv] = s;

        // V: load row into registers, store transposed (all lanes write a
        // contiguous 256B run per d-step -> conflict-free)
        f32x4 vbuf4[8];
        const f32x4* vp = (const f32x4*)(vb + headOff + (size_t)s * D_);
#pragma unroll
        for (int i = 0; i < 8; ++i) vbuf4[i] = vp[i];
        const __bf16* vbuf = (const __bf16*)vbuf4;
#pragma unroll
        for (int d = 0; d < 64; ++d) Vt[d][rkv] = vbuf[d];

        if (rkv >= 64) {
            const int qr = rkv - 64;
            // raw Q copy: pure global->LDS move, use CDNA5 async-to-LDS DMA
            // (wave-uniform branch: waves 2..3 fully active -> EXEC all ones)
            unsigned ldsAddr = (unsigned)(size_t)(&Qs[qr][0]);
            unsigned long long gaddr = (unsigned long long)(size_t)qrow;
#pragma unroll
            for (int i = 0; i < 8; ++i) {
                asm volatile("global_load_async_to_lds_b128 %0, %1, off"
                             :: "v"(ldsAddr + i * 16), "v"(gaddr + i * 16)
                             : "memory");
            }
            QIdx[qr] = s;
            QJ[qr]   = j;
        }
    }
    // drain this wave's async LDS writes, then join the block barrier
    asm volatile("s_wait_asynccnt 0x0" ::: "memory");
    __syncthreads();

    const int wave = tid >> 5;
    const int lane = tid & 31;
    const int hi   = lane >> 4;
    const int ln   = lane & 15;

    // ---- phase 1: dots = Q x K^T for this wave's 16 query rows ----
    v8f dacc[8];
#pragma unroll
    for (int nt = 0; nt < 8; ++nt) {
        v8f acc = {};
#pragma unroll
        for (int ks = 0; ks < 2; ++ks) {
            v16bf af = loadA_bf16(&Qs[wave * 16 + ln][0], ks * 32, hi);
            v16bf bf = loadB_rowk(&Ks[nt * 16 + ln][0], ks * 32, hi);
            acc = wmma_bf16(af, bf, acc);
        }
        dacc[nt] = acc;
    }

    // ---- masks (causal then self), in registers ----
#pragma unroll
    for (int nt = 0; nt < 8; ++nt) {
#pragma unroll
        for (int r = 0; r < 8; ++r) {
            int m  = wave * 16 + r + 8 * hi;
            int n  = nt * 16 + ln;
            int qs = QIdx[m], kv = KIdx[n];
            float v = dacc[nt][r];
            v = (qs < kv)  ? -1e9f : v;
            v = (qs == kv) ? -1e5f : v;
            dacc[nt][r] = v;
        }
    }

    // ---- in-wave softmax over 128 keys (rows live on 16-lane halves) ----
    float logit[8];
#pragma unroll
    for (int r = 0; r < 8; ++r) {
        float mx = dacc[0][r];
#pragma unroll
        for (int nt = 1; nt < 8; ++nt) mx = fmaxf(mx, dacc[nt][r]);
#pragma unroll
        for (int off = 1; off < 16; off <<= 1) mx = fmaxf(mx, __shfl_xor(mx, off, 32));
        float sm = 0.f;
#pragma unroll
        for (int nt = 0; nt < 8; ++nt) sm += __expf(dacc[nt][r] - mx);
#pragma unroll
        for (int off = 1; off < 16; off <<= 1) sm += __shfl_xor(sm, off, 32);
        logit[r] = mx + __logf(sm);
    }

#pragma unroll
    for (int nt = 0; nt < 8; ++nt) {
#pragma unroll
        for (int r = 0; r < 8; ++r) {
            int m = wave * 16 + r + 8 * hi;
            int n = nt * 16 + ln;
            Ps[m][n] = (__bf16)__expf(dacc[nt][r] - logit[r]);
        }
    }
    if (ln == 0) {
#pragma unroll
        for (int r = 0; r < 8; ++r) {
            int m  = wave * 16 + r + 8 * hi;
            int j  = QJ[m];
            int rh = j >> 12, s = j & (S_ - 1);
            logitsg[(size_t)(bh * NH_ + rh) * S_ + s] = logit[r];
        }
    }
    __syncthreads();

    // ---- phase 2: out = P x V, scatter to un-sorted [b,h,round,s,d] ----
#pragma unroll
    for (int nt = 0; nt < 4; ++nt) {
        v8f acc = {};
#pragma unroll
        for (int ks = 0; ks < 4; ++ks) {
            v16bf af = loadA_bf16(&Ps[wave * 16 + ln][0], ks * 32, hi);
            // B[k][n] = V[k][n] = Vt[n][k] -> contiguous vector LDS loads
            v16bf bf = loadB_rowk(&Vt[nt * 16 + ln][0], ks * 32, hi);
            acc = wmma_bf16(af, bf, acc);
        }
#pragma unroll
        for (int r = 0; r < 8; ++r) {
            int m  = wave * 16 + r + 8 * hi;
            int j  = QJ[m];
            int rh = j >> 12, s = j & (S_ - 1);
            outr[(((size_t)(bh * NH_ + rh)) * S_ + s) * D_ + nt * 16 + ln] = (__bf16)acc[r];
        }
    }
}

// ---------------------------------------------------------------------------
// K5: combine hash rounds with softmax over logits; write [b,s,h*D] f32
// ---------------------------------------------------------------------------
__global__ void combine(const __bf16* __restrict__ outr,
                        const float* __restrict__ logitsg,
                        float* __restrict__ outF) {
    const size_t gid = (size_t)blockIdx.x * blockDim.x + threadIdx.x;
    const size_t total = (size_t)B_ * S_ * H_ * D_;
    if (gid >= total) return;
    const int d  = (int)(gid & 63);
    const int h  = (int)((gid >> 6) & 15);
    const size_t sg = gid >> 10;          // b*S + s
    const int b  = (int)(sg >> 12);
    const int s  = (int)(sg & (S_ - 1));
    const int bh = b * H_ + h;

    float l[4];
#pragma unroll
    for (int r = 0; r < 4; ++r) l[r] = logitsg[(size_t)(bh * NH_ + r) * S_ + s];
    float mx = fmaxf(fmaxf(l[0], l[1]), fmaxf(l[2], l[3]));
    float e[4];
    float sum = 0.f;
#pragma unroll
    for (int r = 0; r < 4; ++r) { e[r] = __expf(l[r] - mx); sum += e[r]; }
    const float inv = 1.f / sum;

    float acc = 0.f;
#pragma unroll
    for (int r = 0; r < 4; ++r)
        acc += e[r] * (float)outr[(((size_t)(bh * NH_ + r)) * S_ + s) * D_ + d];
    outF[gid] = acc * inv;
}

// ---------------------------------------------------------------------------
extern "C" void kernel_launch(void* const* d_in, const int* in_sizes, int n_in,
                              void* d_out, int out_size, void* d_ws, size_t ws_size,
                              hipStream_t stream) {
    (void)in_sizes; (void)n_in; (void)out_size; (void)ws_size;
    const float* hidden = (const float*)d_in[0];  // [B,S,HID] f32
    const float* wqk    = (const float*)d_in[1];  // [HID, H*D] f32
    const float* wv     = (const float*)d_in[2];  // [HID, H*D] f32
    const float* rot    = (const float*)d_in[3];  // [H,D,NH,64] f32
    float* outF         = (float*)d_out;          // [B,S,H*D] f32

    char* ws = (char*)d_ws;
    size_t off = 0;
    __bf16* qkb = (__bf16*)(ws + off); off += (size_t)B_ * H_ * S_ * D_ * 2;       // 16 MB
    __bf16* vbp = (__bf16*)(ws + off); off += (size_t)B_ * H_ * S_ * D_ * 2;       // 16 MB
    int* buckets = (int*)(ws + off);   off += (size_t)B_ * H_ * TOT_ * 4;          //  2 MB
    int* sidxp   = (int*)(ws + off);   off += (size_t)B_ * H_ * TOT_ * 4;          //  2 MB
    float* logitsg = (float*)(ws + off); off += (size_t)B_ * H_ * TOT_ * 4;        //  2 MB
    __bf16* outrb  = (__bf16*)(ws + off); off += (size_t)B_ * H_ * TOT_ * D_ * 2;  // 64 MB

    proj_gemm<<<dim3(8192, 2, 1), 32, 0, stream>>>(hidden, wqk, wv, qkb, vbp);
    lsh_hash<<<dim3(S_ / 64, H_, B_), 64, 0, stream>>>(qkb, rot, buckets);
    bucket_sort<<<dim3(B_ * H_, 1, 1), 64, 0, stream>>>(buckets, sidxp);
    lsh_attn<<<dim3(NCHUNK_, H_, B_), 128, 0, stream>>>(qkb, vbp, sidxp, outrb, logitsg);
    combine<<<dim3(32768, 1, 1), 256, 0, stream>>>(outrb, logitsg, outF);
}
